// AttentionGNN_52089363366520
// MI455X (gfx1250) — compile-verified, matrix operand order checked by reading, not run on hardware
//
#include <hip/hip_runtime.h>
#include <hip/hip_bf16.h>

typedef __attribute__((ext_vector_type(16))) _Float16 v16h;
typedef __attribute__((ext_vector_type(8)))  _Float16 v8h;
typedef __attribute__((ext_vector_type(8)))  float    v8f;

#define N_NODES   50000
#define N_EDGES   800000
#define N_TOT_E   (N_EDGES + N_NODES)   // + self loops
#define FDIM      128
#define NEG_SLOPE 0.2f
#define LN_EPS    1e-5f

// ---------------- weight convert: W[k][c] f32 -> WhT[c][k] f16 ----------------
__global__ void wconv_kernel(const float* __restrict__ W, _Float16* __restrict__ WhT) {
  int t = blockIdx.x * blockDim.x + threadIdx.x;
  if (t >= FDIM * FDIM) return;
  int c = t >> 7, k = t & 127;
  WhT[t] = (_Float16)W[k * FDIM + c];
}

__global__ void fill_kernel(float* __restrict__ p, float v, int n) {
  int t = blockIdx.x * blockDim.x + threadIdx.x;
  if (t < n) p[t] = v;
}

// ---------------- WMMA GEMM: Y[N,128] = f16(X[N,128]) @ f16(W[128,128]) ----------------
// One wave computes a 16x16 tile; 8 waves/block cover all 128 cols of a 16-row band.
__global__ void __launch_bounds__(256)
gemm_wmma_kernel(const float* __restrict__ X, const _Float16* __restrict__ WhT,
                 float* __restrict__ Y) {
  const int wave  = threadIdx.x >> 5;          // column tile 0..7
  const int lane  = threadIdx.x & 31;
  const int mrow  = lane & 15;
  const int kbase = (lane < 16) ? 0 : 8;       // ISA 16-bit A/B 16x32 layout
  const int row0  = blockIdx.x * 16;

  const float*    xp = X   + (size_t)(row0 + mrow) * FDIM;        // A: row per lane
  const _Float16* wp = WhT + (size_t)(wave * 16 + mrow) * FDIM;   // B: col per lane

  v8f acc = {};
#pragma unroll
  for (int kk = 0; kk < 4; ++kk) {             // K = 128 = 4 x 32
    const int k0 = kk * 32 + kbase;
    const float4 f0 = *(const float4*)(xp + k0);
    const float4 f1 = *(const float4*)(xp + k0 + 4);
    const float4 f2 = *(const float4*)(xp + k0 + 16);
    const float4 f3 = *(const float4*)(xp + k0 + 20);
    v16h a;
    a[0]=(_Float16)f0.x;  a[1]=(_Float16)f0.y;  a[2]=(_Float16)f0.z;  a[3]=(_Float16)f0.w;
    a[4]=(_Float16)f1.x;  a[5]=(_Float16)f1.y;  a[6]=(_Float16)f1.z;  a[7]=(_Float16)f1.w;
    a[8]=(_Float16)f2.x;  a[9]=(_Float16)f2.y;  a[10]=(_Float16)f2.z; a[11]=(_Float16)f2.w;
    a[12]=(_Float16)f3.x; a[13]=(_Float16)f3.y; a[14]=(_Float16)f3.z; a[15]=(_Float16)f3.w;
    v8h b0 = *(const v8h*)(wp + k0);
    v8h b1 = *(const v8h*)(wp + k0 + 16);
    v16h b;
#pragma unroll
    for (int i = 0; i < 8; ++i) { b[i] = b0[i]; b[8 + i] = b1[i]; }
    acc = __builtin_amdgcn_wmma_f32_16x16x32_f16(false, a, false, b,
                                                 (short)0, acc, false, false);
  }
  const int mbase = (lane < 16) ? 0 : 8;       // C/D layout: VGPR r -> row mbase+r
  float* yp = Y + (size_t)(row0 + mbase) * FDIM + wave * 16 + mrow;
#pragma unroll
  for (int r = 0; r < 8; ++r) yp[(size_t)r * FDIM] = acc[r];
}

// ---------------- per-node attention logits: es/ed [N,H] ----------------
__global__ void esed_kernel(const float* __restrict__ feat,
                            const float* __restrict__ asrc, const float* __restrict__ adst,
                            float* __restrict__ es, float* __restrict__ ed,
                            int H, int C) {
  int t = blockIdx.x * blockDim.x + threadIdx.x;
  if (t >= N_NODES * H) return;
  int n = t / H, h = t - (t / H) * H;
  const float* fp = feat + (size_t)n * FDIM + h * C;
  const float* ap = asrc + h * C;
  const float* bp = adst + h * C;
  float s = 0.f, d = 0.f;
  for (int c = 0; c < C; ++c) { float v = fp[c]; s += v * ap[c]; d += v * bp[c]; }
  es[t] = s; ed[t] = d;
}

// ---------------- edge helpers ----------------
__device__ __forceinline__ void edge_sd(const long long* __restrict__ ei, int e,
                                        int& s, int& d) {
  if (e < N_EDGES) { s = (int)ei[e]; d = (int)ei[N_EDGES + e]; }
  else             { s = e - N_EDGES; d = s; }     // self loop
}

__device__ __forceinline__ void atomic_max_f32(float* addr, float v) {
  if (v >= 0.f) atomicMax((int*)addr, __float_as_int(v));
  else          atomicMin((unsigned int*)addr, __float_as_uint(v));
}

// pass A: e = leaky_relu(es[src]+ed[dst]); segment max over dst
__global__ void edge_max_kernel(const long long* __restrict__ ei,
                                const float* __restrict__ es, const float* __restrict__ ed,
                                float* __restrict__ ebuf, float* __restrict__ m, int H) {
  int t = blockIdx.x * blockDim.x + threadIdx.x;
  if (t >= N_TOT_E * H) return;
  int e = t / H, h = t - (t / H) * H;
  int s, d; edge_sd(ei, e, s, d);
  float v = es[s * H + h] + ed[d * H + h];
  v = (v > 0.f) ? v : NEG_SLOPE * v;
  ebuf[t] = v;
  atomic_max_f32(&m[d * H + h], v);
}

// pass B: a = exp(e - m[dst]); segment sum -> den
__global__ void edge_exp_kernel(const long long* __restrict__ ei,
                                float* __restrict__ ebuf, const float* __restrict__ m,
                                float* __restrict__ den, int H) {
  int t = blockIdx.x * blockDim.x + threadIdx.x;
  if (t >= N_TOT_E * H) return;
  int e = t / H, h = t - (t / H) * H;
  int s, d; edge_sd(ei, e, s, d);
  float a = __expf(ebuf[t] - m[d * H + h]);
  ebuf[t] = a;
  atomicAdd(&den[d * H + h], a);
}

// pass C: out[dst,c] += (a/den[dst,h]) * feat[src,c]; one block per edge, 128 ch
__global__ void __launch_bounds__(128)
edge_agg_kernel(const long long* __restrict__ ei,
                const float* __restrict__ ebuf, const float* __restrict__ den,
                const float* __restrict__ feat, float* __restrict__ out,
                int H, int logC) {
  int e = blockIdx.x;
  int c = threadIdx.x;
  int s, d; edge_sd(ei, e, s, d);
  int h = c >> logC;
  float alpha = ebuf[e * H + h] / den[d * H + h];
  atomicAdd(&out[(size_t)d * FDIM + c], alpha * feat[(size_t)s * FDIM + c]);
}

// ---------------- bias + LayerNorm + ReLU (one wave per node) ----------------
__global__ void __launch_bounds__(256)
ln_relu_kernel(const float* __restrict__ in, const float* __restrict__ bias,
               const float* __restrict__ g, const float* __restrict__ be,
               float* __restrict__ out) {
  int lane = threadIdx.x & 31;
  int node = blockIdx.x * (blockDim.x >> 5) + (threadIdx.x >> 5);
  if (node >= N_NODES) return;
  const float* ip = in + (size_t)node * FDIM;
  float v[4]; float sum = 0.f;
#pragma unroll
  for (int i = 0; i < 4; ++i) { int c = lane + 32 * i; v[i] = ip[c] + bias[c]; sum += v[i]; }
#pragma unroll
  for (int o = 16; o > 0; o >>= 1) sum += __shfl_xor(sum, o, 32);
  float mu = sum * (1.0f / FDIM);
  float var = 0.f;
#pragma unroll
  for (int i = 0; i < 4; ++i) { float t = v[i] - mu; var += t * t; }
#pragma unroll
  for (int o = 16; o > 0; o >>= 1) var += __shfl_xor(var, o, 32);
  float rs = rsqrtf(var * (1.0f / FDIM) + LN_EPS);
  float* op = out + (size_t)node * FDIM;
#pragma unroll
  for (int i = 0; i < 4; ++i) {
    int c = lane + 32 * i;
    float r = (v[i] - mu) * rs * g[c] + be[c];
    op[c] = r > 0.f ? r : 0.f;
  }
}

// ---------------- host: full two-layer pipeline ----------------
extern "C" void kernel_launch(void* const* d_in, const int* in_sizes, int n_in,
                              void* d_out, int out_size, void* d_ws, size_t ws_size,
                              hipStream_t stream) {
  const float*     x   = (const float*)d_in[0];
  const long long* ei  = (const long long*)d_in[1];
  const float*     W1  = (const float*)d_in[2];
  const float*     as1 = (const float*)d_in[3];
  const float*     ad1 = (const float*)d_in[4];
  const float*     b1  = (const float*)d_in[5];
  const float*     W2  = (const float*)d_in[6];
  const float*     as2 = (const float*)d_in[7];
  const float*     ad2 = (const float*)d_in[8];
  const float*     b2  = (const float*)d_in[9];
  const float*     g1  = (const float*)d_in[10];
  const float*     be1 = (const float*)d_in[11];
  const float*     g2  = (const float*)d_in[12];
  const float*     be2 = (const float*)d_in[13];
  float* out = (float*)d_out;

  const size_t NF = (size_t)N_NODES * FDIM;
  char* ws = (char*)d_ws;
  size_t off = 0;
  float* h1   = (float*)(ws + off); off += NF * 4;                 // layer1 features; reused as out2
  float* agg1 = (float*)(ws + off); off += NF * 4;                 // layer1 aggregation; reused as t2
  float* h2   = (float*)(ws + off); off += NF * 4;                 // post-LN layer1 output
  float* ebuf = (float*)(ws + off); off += (size_t)N_TOT_E * 4 * 4;
  float* es   = (float*)(ws + off); off += (size_t)N_NODES * 4 * 4;
  float* ed   = (float*)(ws + off); off += (size_t)N_NODES * 4 * 4;
  float* mbuf = (float*)(ws + off); off += (size_t)N_NODES * 4 * 4;
  float* den  = (float*)(ws + off); off += (size_t)N_NODES * 4 * 4;
  _Float16* W1h = (_Float16*)(ws + off); off += FDIM * FDIM * 2;
  _Float16* W2h = (_Float16*)(ws + off); off += FDIM * FDIM * 2;
  float* out2 = h1;      // reuse
  float* t2   = agg1;    // reuse

  const int B = 256;
  auto cdiv = [](int a, int b) { return (a + b - 1) / b; };

  // ---- layer 1 (H=4, C=32) ----
  wconv_kernel<<<cdiv(FDIM * FDIM, B), B, 0, stream>>>(W1, W1h);
  gemm_wmma_kernel<<<N_NODES / 16, B, 0, stream>>>(x, W1h, h1);
  esed_kernel<<<cdiv(N_NODES * 4, B), B, 0, stream>>>(h1, as1, ad1, es, ed, 4, 32);
  fill_kernel<<<cdiv(N_NODES * 4, B), B, 0, stream>>>(mbuf, -3.0e38f, N_NODES * 4);
  fill_kernel<<<cdiv(N_NODES * 4, B), B, 0, stream>>>(den, 0.f, N_NODES * 4);
  fill_kernel<<<cdiv((int)NF, B), B, 0, stream>>>(agg1, 0.f, (int)NF);
  edge_max_kernel<<<cdiv(N_TOT_E * 4, B), B, 0, stream>>>(ei, es, ed, ebuf, mbuf, 4);
  edge_exp_kernel<<<cdiv(N_TOT_E * 4, B), B, 0, stream>>>(ei, ebuf, mbuf, den, 4);
  edge_agg_kernel<<<N_TOT_E, 128, 0, stream>>>(ei, ebuf, den, h1, agg1, 4, 5);
  ln_relu_kernel<<<cdiv(N_NODES, 8), B, 0, stream>>>(agg1, b1, g1, be1, h2);

  // ---- layer 2 (H=1, C=128) ----
  wconv_kernel<<<cdiv(FDIM * FDIM, B), B, 0, stream>>>(W2, W2h);
  gemm_wmma_kernel<<<N_NODES / 16, B, 0, stream>>>(h2, W2h, t2);
  esed_kernel<<<cdiv(N_NODES, B), B, 0, stream>>>(t2, as2, ad2, es, ed, 1, 128);
  fill_kernel<<<cdiv(N_NODES, B), B, 0, stream>>>(mbuf, -3.0e38f, N_NODES);
  fill_kernel<<<cdiv(N_NODES, B), B, 0, stream>>>(den, 0.f, N_NODES);
  fill_kernel<<<cdiv((int)NF, B), B, 0, stream>>>(out2, 0.f, (int)NF);
  edge_max_kernel<<<cdiv(N_TOT_E, B), B, 0, stream>>>(ei, es, ed, ebuf, mbuf, 1);
  edge_exp_kernel<<<cdiv(N_TOT_E, B), B, 0, stream>>>(ei, ebuf, mbuf, den, 1);
  edge_agg_kernel<<<N_TOT_E, 128, 0, stream>>>(ei, ebuf, den, t2, out2, 1, 7);
  ln_relu_kernel<<<cdiv(N_NODES, 8), B, 0, stream>>>(out2, b2, g2, be2, out);
}
